// RoPEAttnLayer_48567490183543
// MI455X (gfx1250) — matrix operand order repaired and checked by reading, not compiled, not run on hardware
//
#include <hip/hip_runtime.h>
#include <hip/hip_bf16.h>

// ---------------------------------------------------------------------------
// RoPE attention block for MI455X (gfx1250): wave32, WMMA bf16, async
// global->LDS staging (ASYNCcnt) with double-buffered LDS tiles.
// ---------------------------------------------------------------------------

typedef __bf16 bf16;
typedef __attribute__((ext_vector_type(16))) __bf16 v16bf;
typedef __attribute__((ext_vector_type(8)))  __bf16 v8bf;
typedef __attribute__((ext_vector_type(8)))  float  v8f;

#define BB 2
#define TT 2048
#define CC 1024
#define HH 16
#define DH 64
#define BT (BB*TT)       // 4096
#define C3 (3*CC)        // 3072

#define WMMA_BF16(a,b,c) \
  __builtin_amdgcn_wmma_f32_16x16x32_bf16(false,(a),false,(b),(short)0,(c),false,false)

// ----- async global->LDS path (gfx1250), with graceful fallback -------------
#if defined(__has_builtin)
# if __has_builtin(__builtin_amdgcn_global_load_async_to_lds_b128) && \
     __has_builtin(__builtin_amdgcn_s_wait_asynccnt)
#  define HAVE_ASYNC_LDS 1
# endif
#endif
#ifndef HAVE_ASYNC_LDS
# define HAVE_ASYNC_LDS 0
#endif

#if HAVE_ASYNC_LDS
// builtin signature (from clang diagnostics): (int4 AS1*, int4 AS3*, imm, imm)
typedef int v4i __attribute__((vector_size(4 * sizeof(int))));
typedef __attribute__((address_space(1))) v4i glb_i4;
typedef __attribute__((address_space(3))) v4i lds_i4;
// copy 32 bytes (one thread) global -> LDS, no VGPR round-trip
#define ASYNC_CP32(gp, lp) do {                                              \
    __builtin_amdgcn_global_load_async_to_lds_b128((glb_i4*)(gp),            \
                                                   (lds_i4*)(lp), 0, 0);     \
    __builtin_amdgcn_global_load_async_to_lds_b128((glb_i4*)((gp) + 8),      \
                                                   (lds_i4*)((lp) + 8), 0, 0);\
  } while (0)
#define WAIT_ASYNC() __builtin_amdgcn_s_wait_asynccnt(0)
#else
#define ASYNC_CP32(gp, lp) (*(v16bf*)(lp) = *(const v16bf*)(gp))
#define WAIT_ASYNC() ((void)0)
#endif

static __device__ __forceinline__ v16bf join8(v8bf lo, v8bf hi) {
  return __builtin_shufflevector(lo, hi, 0,1,2,3,4,5,6,7,8,9,10,11,12,13,14,15);
}

// ---------------------------------------------------------------------------
// fp32 -> bf16 conversion (4 elements / thread)
// ---------------------------------------------------------------------------
__global__ __launch_bounds__(256) void cvt_bf16_kernel(const float* __restrict__ in,
                                                       bf16* __restrict__ out, int n) {
  int i = (blockIdx.x * 256 + threadIdx.x) * 4;
  if (i < n) {
    float4 f = *(const float4*)(in + i);
    out[i]   = (bf16)f.x; out[i+1] = (bf16)f.y;
    out[i+2] = (bf16)f.z; out[i+3] = (bf16)f.w;
  }
}

// ---------------------------------------------------------------------------
// bf16 GEMM: C[M,N] = A[M,K] * Bw^T  (Bw stored [N][K], K contiguous per col)
// fp32 accumulate, optional fp32 residual. Block tile 128x128, K-step 32,
// double-buffered LDS staged via async global->LDS DMA.
// 256 threads = 8 waves in 4(m) x 2(n); each wave -> 32x64 of output.
// ---------------------------------------------------------------------------
__global__ __launch_bounds__(256) void gemm_bf16_kernel(
    const bf16* __restrict__ A, const bf16* __restrict__ Bw,
    const float* __restrict__ resid, float* __restrict__ Cout,
    int M, int N, int K)
{
  __shared__ __align__(64) bf16 As[2][128][32];
  __shared__ __align__(64) bf16 Bs[2][128][32];

  const int tid   = threadIdx.x;
  const int lane  = tid & 31;
  const int wave  = tid >> 5;
  const int wm    = wave >> 1;          // 0..3
  const int wn    = wave & 1;           // 0..1
  const int m0    = blockIdx.y * 128;
  const int n0    = blockIdx.x * 128;
  const int lrow  = lane & 15;
  const int lhi   = lane >> 4;          // 0/1
  const int kbase = lhi * 8;

  const int srow  = tid >> 1;           // 0..127
  const int shalf = (tid & 1) * 16;     // 0/16

  v8f acc[2][4];
  #pragma unroll
  for (int i = 0; i < 2; ++i)
    #pragma unroll
    for (int j = 0; j < 4; ++j) acc[i][j] = {};

  auto stage = [&](int nb, int k0s) {
    const bf16* ga = A  + (size_t)(m0 + srow) * K + k0s + shalf;
    const bf16* gb = Bw + (size_t)(n0 + srow) * K + k0s + shalf;
    ASYNC_CP32(ga, &As[nb][srow][shalf]);
    ASYNC_CP32(gb, &Bs[nb][srow][shalf]);
  };

  stage(0, 0);
  WAIT_ASYNC();
  __syncthreads();

  int cur = 0;
  for (int k0 = 0; k0 < K; k0 += 32) {
    if (k0 + 32 < K) stage(cur ^ 1, k0 + 32);   // DMA next tile while we compute

    v16bf af[2];
    #pragma unroll
    for (int ms = 0; ms < 2; ++ms) {
      const int r = wm * 32 + ms * 16 + lrow;
      af[ms] = join8(*(const v8bf*)&As[cur][r][kbase],
                     *(const v8bf*)&As[cur][r][kbase + 16]);
    }
    #pragma unroll
    for (int ns = 0; ns < 4; ++ns) {
      const int c = wn * 64 + ns * 16 + lrow;
      v16bf bfv = join8(*(const v8bf*)&Bs[cur][c][lhi * 8], *(const v8bf*)&Bs[cur][c][lhi * 8 + 16]);
      acc[0][ns] = WMMA_BF16(af[0], bfv, acc[0][ns]);
      acc[1][ns] = WMMA_BF16(af[1], bfv, acc[1][ns]);
    }

    WAIT_ASYNC();       // my async stage of next tile is done
    __syncthreads();    // everyone done reading cur + done staging cur^1
    cur ^= 1;
  }

  // epilogue: D layout -> (row = r + 8*lhi, col = lane&15) per 16x16 tile
  #pragma unroll
  for (int ms = 0; ms < 2; ++ms)
    #pragma unroll
    for (int ns = 0; ns < 4; ++ns)
      #pragma unroll
      for (int r = 0; r < 8; ++r) {
        int row = m0 + wm * 32 + ms * 16 + r + lhi * 8;
        int col = n0 + wn * 64 + ns * 16 + lrow;
        float val = acc[ms][ns][r];
        if (resid) val += resid[(size_t)row * N + col];
        Cout[(size_t)row * N + col] = val;
      }
}

// ---------------------------------------------------------------------------
// Split qkv (fp32 [BT][3C]) into bf16 q,k,v and RoPE'd q_rope,k_rope, each
// laid out [B,H,T,64]. One thread per rotation pair (2 elements).
// ---------------------------------------------------------------------------
__global__ __launch_bounds__(256) void rope_pack_kernel(
    const float* __restrict__ qkv,
    bf16* __restrict__ q,  bf16* __restrict__ k,  bf16* __restrict__ v,
    bf16* __restrict__ qr, bf16* __restrict__ kr)
{
  int idx = blockIdx.x * 256 + threadIdx.x;   // < B*T*H*32
  int j = idx & 31;
  int h = (idx >> 5) & 15;
  int t = (idx >> 9) & 2047;
  int b = idx >> 20;

  size_t src = ((size_t)(b * TT + t)) * C3 + h * DH + 2 * j;
  float q0 = qkv[src],        q1 = qkv[src + 1];
  float k0 = qkv[src + CC],   k1 = qkv[src + CC + 1];
  float v0 = qkv[src + 2*CC], v1 = qkv[src + 2*CC + 1];

  float theta = __powf(1000.0f, -2.0f * (float)(j + 1) / 64.0f);
  float arg = (float)(t + 1) * theta;
  float s = __sinf(arg), c = __cosf(arg);

  size_t dst = (((size_t)(b * HH + h)) * TT + t) * DH + 2 * j;
  q[dst] = (bf16)q0;  q[dst + 1] = (bf16)q1;
  k[dst] = (bf16)k0;  k[dst + 1] = (bf16)k1;
  v[dst] = (bf16)v0;  v[dst + 1] = (bf16)v1;
  qr[dst] = (bf16)(q0 * c - q1 * s);  qr[dst + 1] = (bf16)(q1 * c + q0 * s);
  kr[dst] = (bf16)(k0 * c - k1 * s);  kr[dst + 1] = (bf16)(k1 * c + k0 * s);
}

// ---------------------------------------------------------------------------
// Attention: num = exp(Qr Kr^T * s) V ; den = rowsum exp(Q K^T * s); y=num/den
// Grid (T/64, B*H); 4 waves/WG, each wave owns 16 q rows. K/Kr async-DMA'd
// into double-buffered LDS; V transposed through VGPRs into LDS. Denominator
// accumulated by WMMA against an all-ones B matrix so the final divide is
// lane-local (identical D-register row mapping).
// ---------------------------------------------------------------------------
__global__ __launch_bounds__(128) void attn_kernel(
    const bf16* __restrict__ q,  const bf16* __restrict__ k,  const bf16* __restrict__ v,
    const bf16* __restrict__ qr, const bf16* __restrict__ kr,
    bf16* __restrict__ yb)
{
  __shared__ __align__(64) bf16 Ks [2][32][64];
  __shared__ __align__(64) bf16 Krs[2][32][64];
  __shared__ __align__(64) bf16 Vts[2][64][32];   // V transposed: [d][key]
  __shared__ __align__(64) bf16 Pn [4][16][32];   // per-wave exp(rope scores)
  __shared__ __align__(64) bf16 Pd [4][16][32];   // per-wave exp(plain scores)

  const int tid   = threadIdx.x;
  const int lane  = tid & 31;
  const int wave  = tid >> 5;           // 0..3
  const int lrow  = lane & 15;
  const int lhi   = lane >> 4;
  const int kbase = lhi * 8;
  const int bh    = blockIdx.y;
  const int b     = bh >> 4;
  const int h     = bh & 15;
  const int q0row = blockIdx.x * 64 + wave * 16;
  const size_t headbase = (size_t)bh * TT * DH;

  // Q fragments (A layout: lanes 0-15 K 0..7/16..23, lanes 16-31 K 8..15/24..31)
  v16bf aq[2], aqr[2];
  {
    const bf16* qp  = q  + headbase + (size_t)(q0row + lrow) * DH;
    const bf16* qrp = qr + headbase + (size_t)(q0row + lrow) * DH;
    #pragma unroll
    for (int d0 = 0; d0 < 2; ++d0) {
      aq[d0]  = join8(*(const v8bf*)(qp  + d0*32 + kbase), *(const v8bf*)(qp  + d0*32 + kbase + 16));
      aqr[d0] = join8(*(const v8bf*)(qrp + d0*32 + kbase), *(const v8bf*)(qrp + d0*32 + kbase + 16));
    }
  }
  v16bf ones;
  #pragma unroll
  for (int i = 0; i < 16; ++i) ones[i] = (bf16)1.0f;

  v8f accY[4]; v8f accD = {};
  #pragma unroll
  for (int i = 0; i < 4; ++i) accY[i] = {};

  const int crow = tid >> 2;          // 0..31 (key within chunk)
  const int cch  = (tid & 3) * 16;    // d chunk

  auto stage = [&](int nb, int kcs) {
    const size_t g = headbase + (size_t)(kcs + crow) * DH + cch;
    ASYNC_CP32(k  + g, &Ks [nb][crow][cch]);
    ASYNC_CP32(kr + g, &Krs[nb][crow][cch]);
    v16bf tv = *(const v16bf*)(v + g);
    #pragma unroll
    for (int i = 0; i < 16; ++i) Vts[nb][cch + i][crow] = tv[i];
    if (kcs + 32 < TT)   // prefetch next V tile (global_prefetch_b8)
      __builtin_prefetch(v + g + (size_t)32 * DH, 0, 1);
  };

  stage(0, 0);
  WAIT_ASYNC();
  __syncthreads();

  int cur = 0;
  for (int kc = 0; kc < TT; kc += 32) {
    if (kc + 32 < TT) stage(cur ^ 1, kc + 32);  // DMA next tile during compute

    #pragma unroll
    for (int ns = 0; ns < 2; ++ns) {
      const int key = ns * 16 + lrow;
      // rope scores -> numerator P
      v8f s = {};
      s = WMMA_BF16(aqr[0], join8(*(const v8bf*)&Krs[cur][key][lhi * 8], *(const v8bf*)&Krs[cur][key][lhi * 8 + 16]),      s);
      s = WMMA_BF16(aqr[1], join8(*(const v8bf*)&Krs[cur][key][32 + lhi * 8], *(const v8bf*)&Krs[cur][key][48 + lhi * 8]), s);
      #pragma unroll
      for (int r = 0; r < 8; ++r)
        Pn[wave][r + lhi * 8][ns * 16 + lrow] = (bf16)__expf(s[r] * 0.125f);
      // plain scores -> denominator P
      v8f sp = {};
      sp = WMMA_BF16(aq[0], join8(*(const v8bf*)&Ks[cur][key][lhi * 8], *(const v8bf*)&Ks[cur][key][lhi * 8 + 16]),      sp);
      sp = WMMA_BF16(aq[1], join8(*(const v8bf*)&Ks[cur][key][32 + lhi * 8], *(const v8bf*)&Ks[cur][key][48 + lhi * 8]), sp);
      #pragma unroll
      for (int r = 0; r < 8; ++r)
        Pd[wave][r + lhi * 8][ns * 16 + lrow] = (bf16)__expf(sp[r] * 0.125f);
    }

    // P as A-fragments (wave-private LDS slice; same-wave DS ops are ordered)
    v16bf apn = join8(*(const v8bf*)&Pn[wave][lrow][kbase],
                      *(const v8bf*)&Pn[wave][lrow][kbase + 16]);
    v16bf apd = join8(*(const v8bf*)&Pd[wave][lrow][kbase],
                      *(const v8bf*)&Pd[wave][lrow][kbase + 16]);

    accD = WMMA_BF16(apd, ones, accD);   // rowsum via ones-matrix WMMA
    #pragma unroll
    for (int dt = 0; dt < 4; ++dt) {
      v16bf bv = join8(*(const v8bf*)&Vts[cur][dt * 16 + lrow][lhi * 8], *(const v8bf*)&Vts[cur][dt * 16 + lrow][lhi * 8 + 16]);
      accY[dt] = WMMA_BF16(apn, bv, accY[dt]);
    }

    WAIT_ASYNC();
    __syncthreads();
    cur ^= 1;
  }

  // y[b, t, h*64 + d]; den has identical D-row mapping -> lane-local divide
  #pragma unroll
  for (int dt = 0; dt < 4; ++dt)
    #pragma unroll
    for (int r = 0; r < 8; ++r) {
      int trow = q0row + r + lhi * 8;
      int d    = dt * 16 + lrow;
      float val = accY[dt][r] / accD[r];
      yb[((size_t)(b * TT + trow)) * CC + h * DH + d] = (bf16)val;
    }
}

// ---------------------------------------------------------------------------
// LayerNorm over rows of h (fp32), 256 threads / row, 4 elems / thread.
// ---------------------------------------------------------------------------
__global__ __launch_bounds__(256) void ln_kernel(
    const float* __restrict__ hbuf, const float* __restrict__ gamma,
    const float* __restrict__ beta, float* __restrict__ out)
{
  __shared__ float red[256];
  const int row = blockIdx.x;
  const int tid = threadIdx.x;
  const float4 hv = *(const float4*)(hbuf + (size_t)row * CC + tid * 4);

  red[tid] = hv.x + hv.y + hv.z + hv.w;
  __syncthreads();
  for (int st = 128; st > 0; st >>= 1) { if (tid < st) red[tid] += red[tid + st]; __syncthreads(); }
  float mu = red[0] * (1.0f / CC);
  __syncthreads();

  float dx = hv.x - mu, dy = hv.y - mu, dz = hv.z - mu, dw = hv.w - mu;
  red[tid] = dx * dx + dy * dy + dz * dz + dw * dw;
  __syncthreads();
  for (int st = 128; st > 0; st >>= 1) { if (tid < st) red[tid] += red[tid + st]; __syncthreads(); }
  float rstd = rsqrtf(red[0] * (1.0f / CC) + 1e-5f);

  float4 g  = *(const float4*)(gamma + tid * 4);
  float4 bt = *(const float4*)(beta  + tid * 4);
  float4 o;
  o.x = dx * rstd * g.x + bt.x;
  o.y = dy * rstd * g.y + bt.y;
  o.z = dz * rstd * g.z + bt.z;
  o.w = dw * rstd * g.w + bt.w;
  *(float4*)(out + (size_t)row * CC + tid * 4) = o;
}

// ---------------------------------------------------------------------------
extern "C" void kernel_launch(void* const* d_in, const int* in_sizes, int n_in,
                              void* d_out, int out_size, void* d_ws, size_t ws_size,
                              hipStream_t stream)
{
  (void)in_sizes; (void)n_in; (void)out_size; (void)ws_size;
  const float* x     = (const float*)d_in[0];
  const float* w_in  = (const float*)d_in[1];
  const float* w_out = (const float*)d_in[2];
  const float* gamma = (const float*)d_in[3];
  const float* beta  = (const float*)d_in[4];
  float* out = (float*)d_out;

  char* ws = (char*)d_ws;
  size_t off = 0;
  auto carve = [&](size_t bytes) -> char* {
    char* p = ws + off;
    off += (bytes + 255) & ~(size_t)255;
    return p;
  };
  bf16*  xb    = (bf16*) carve((size_t)BT * CC * 2);
  bf16*  winb  = (bf16*) carve((size_t)C3 * CC * 2);
  bf16*  woutb = (bf16*) carve((size_t)CC * CC * 2);
  float* qkv   = (float*)carve((size_t)BT * C3 * 4);
  bf16*  qb    = (bf16*) carve((size_t)BT * CC * 2);
  bf16*  kb    = (bf16*) carve((size_t)BT * CC * 2);
  bf16*  vb    = (bf16*) carve((size_t)BT * CC * 2);
  bf16*  qrb   = (bf16*) carve((size_t)BT * CC * 2);
  bf16*  krb   = (bf16*) carve((size_t)BT * CC * 2);
  bf16*  yb    = (bf16*) carve((size_t)BT * CC * 2);
  float* hbuf  = (float*)carve((size_t)BT * CC * 4);

  // 1) fp32 -> bf16 conversions
  cvt_bf16_kernel<<<(BT * CC) / 4 / 256, 256, 0, stream>>>(x,     xb,    BT * CC);
  cvt_bf16_kernel<<<(C3 * CC) / 4 / 256, 256, 0, stream>>>(w_in,  winb,  C3 * CC);
  cvt_bf16_kernel<<<(CC * CC) / 4 / 256, 256, 0, stream>>>(w_out, woutb, CC * CC);

  // 2) qkv = x @ w_in^T   (M=4096, N=3072, K=1024)
  gemm_bf16_kernel<<<dim3(C3 / 128, BT / 128), 256, 0, stream>>>(
      xb, winb, nullptr, qkv, BT, C3, CC);

  // 3) split + RoPE + pack to [B,H,T,64] bf16
  rope_pack_kernel<<<(BB * TT * HH * 32) / 256, 256, 0, stream>>>(
      qkv, qb, kb, vb, qrb, krb);

  // 4) attention (grid: q-blocks x (B*H))
  attn_kernel<<<dim3(TT / 64, BB * HH), 128, 0, stream>>>(
      qb, kb, vb, qrb, krb, yb);

  // 5) h = y @ w_out^T + x   (M=4096, N=1024, K=1024)
  gemm_bf16_kernel<<<dim3(CC / 128, BT / 128), 256, 0, stream>>>(
      yb, woutb, x, hbuf, BT, CC, CC);

  // 6) LayerNorm -> out
  ln_kernel<<<BT, 256, 0, stream>>>(hbuf, gamma, beta, out);
}
